// GSLSAModule_8770323218767
// MI455X (gfx1250) — matrix operand order repaired and checked by reading, not compiled, not run on hardware
//
#include <hip/hip_runtime.h>
#include <hip/hip_bf16.h>
#include <math.h>

// ---------- problem constants ----------
#define NWAY   5
#define NSHOT  5
#define CDIM   640
#define HWDIM  576          // 24*24
#define PERIMG (CDIM*HWDIM) // 368640
#define NIMG   (NWAY*NSHOT) // 25
#define ROWS   (NSHOT*HWDIM)// 2880
#define TOPK   2304         // 2880*0.8

typedef __attribute__((ext_vector_type(16))) _Float16 v16h;
typedef __attribute__((ext_vector_type(8)))  _Float16 v8h;
typedef __attribute__((ext_vector_type(8)))  float    v8f;

// convert 8 strided f32 -> packed v8h (one ds_store_b128 at the call site)
__device__ __forceinline__ v8h cvt8(const float* __restrict__ src, int stride) {
  v8h o;
  #pragma unroll
  for (int j = 0; j < 8; ++j) o[j] = (_Float16)src[(size_t)j * stride];
  return o;
}

// =====================================================================
// 1x1 conv as GEMM:  Out[b] = W(640x640) * X[b](640x576) + bias (+R)
// Block: 256 thr (8 waves), tile 128(M) x 64(N), K-step 32, f16 WMMA.
// A frag (16x32 f16): lanes 0-15 hold K {0..7,16..23}, lanes 16-31
// hold K {8..15,24..31} (ISA 7.12.2).  B frag: lane col = lane%16,
// same K order as A.
// =====================================================================
template <bool HASR>
__global__ __launch_bounds__(256) void k_conv1x1(
    const float* __restrict__ Wm, const float* __restrict__ bias,
    const float* __restrict__ X,  const float* __restrict__ R,
    float* __restrict__ Out)
{
  const int b  = blockIdx.z;
  const int m0 = blockIdx.x * 128;
  const int n0 = blockIdx.y * 64;
  const float* Xb = X + (size_t)b * PERIMG;
  float* Ob = Out + (size_t)b * PERIMG;

  __shared__ _Float16 As[128][40]; // [m][k], row stride 80B (16B aligned)
  __shared__ _Float16 Bs[64][40];  // [n][k] (transposed store)

  const int tid  = threadIdx.x;
  const int lane = tid & 31;
  const int wv   = tid >> 5;   // 0..7 -> 16 rows each
  const int g    = lane >> 4;  // half-wave group
  const int r    = lane & 15;

  // staging coordinates (fixed per thread)
  const int am = tid >> 1;            // A row
  const int ak = (tid & 1) * 16;      // A k-base
  const int bn = tid & 63;            // B column
  const int bk = (tid >> 6) * 8;      // B k-base (0,8,16,24)

  v8f acc0 = {}, acc1 = {}, acc2 = {}, acc3 = {};

  for (int k0 = 0; k0 < CDIM; k0 += 32) {
    __syncthreads();
    { // stage A: 128x32, 2 threads per row, packed b128 stores
      const float* src = Wm + (size_t)(m0 + am) * CDIM + k0 + ak;
      *(v8h*)&As[am][ak]     = cvt8(src, 1);
      *(v8h*)&As[am][ak + 8] = cvt8(src + 8, 1);
    }
    { // stage B: 32x64 -> Bs[n][k], 8 contiguous k per thread, one b128
      *(v8h*)&Bs[bn][bk] =
          cvt8(&Xb[(size_t)(k0 + bk) * HWDIM + n0 + bn], HWDIM);
    }
    if (k0 + 32 < CDIM) { // prefetch next K-step source lines
      __builtin_prefetch(Wm + (size_t)(m0 + am) * CDIM + k0 + 32 + ak, 0, 1);
      __builtin_prefetch(&Xb[(size_t)(k0 + 32 + bk) * HWDIM + n0 + bn], 0, 1);
    }
    __syncthreads();

    union { v16h v; v8h h[2]; } af;
    {
      const _Float16* ap = &As[wv * 16 + r][0];
      af.h[0] = *(const v8h*)(ap + g * 8);
      af.h[1] = *(const v8h*)(ap + 16 + g * 8);
    }
    #pragma unroll
    for (int t = 0; t < 4; ++t) {
      union { v16h v; v8h h[2]; } bf;
      const _Float16* bp = &Bs[t * 16 + r][0];
      bf.h[0] = *(const v8h*)(bp + g * 8);
      bf.h[1] = *(const v8h*)(bp + 16 + g * 8);
      v8f& acc = (t==0)?acc0:((t==1)?acc1:((t==2)?acc2:acc3));
      acc = __builtin_amdgcn_wmma_f32_16x16x32_f16(
          false, af.v, false, bf.v, (short)0, acc, false, false);
    }
  }

  // branchless epilogue (residual resolved at compile time)
  const int row0 = m0 + wv * 16 + g * 8;
  #pragma unroll
  for (int t = 0; t < 4; ++t) {
    v8f acc = (t==0)?acc0:((t==1)?acc1:((t==2)?acc2:acc3));
    const int col = n0 + t * 16 + r;
    float* op = Ob + (size_t)row0 * HWDIM + col;
    const float* rp = HASR ? (R + (size_t)b * PERIMG + (size_t)row0 * HWDIM + col)
                           : nullptr;
    #pragma unroll
    for (int v = 0; v < 8; ++v) {
      float val = acc[v] + bias[row0 + v];
      if (HASR) val += rp[(size_t)v * HWDIM];
      op[(size_t)v * HWDIM] = val;
    }
  }
}

// =====================================================================
// Fused attention row statistic:
//   xw[n,i] = mean_over_5_chunks( max_over_576_cols( Q[n,i,:] . K[n,:,j] ) )
// Score matrix (5x2880x2880, 166MB) never materialized.
// Block: 64 thr (2 waves), 32 rows/block; Q rows cached in LDS as f16.
// =====================================================================
__global__ __launch_bounds__(64) void k_attn_rowstat(
    const float* __restrict__ XQ, const float* __restrict__ XK,
    float* __restrict__ xw)
{
  const int n  = blockIdx.y;        // way
  const int i0 = blockIdx.x * 32;   // row base (576%32==0 -> single shot)
  const int shotA = i0 / HWDIM;
  const int pA0   = i0 % HWDIM;
  const float* Qb = XQ + (size_t)(n * NSHOT + shotA) * PERIMG;

  __shared__ _Float16 Aq[32][648];  // row stride 1296B (16B aligned)
  __shared__ _Float16 Bs[64][40];

  const int tid  = threadIdx.x;
  const int lane = tid & 31;
  const int wv   = tid >> 5;   // 0..1
  const int g    = lane >> 4;
  const int r    = lane & 15;

  { // stage Q rows: packed b128 stores, 8 contiguous c per store
    const int ii = tid & 31;
    const int h8 = (tid >> 5) * 8;
    for (int cb = 0; cb < CDIM; cb += 16) {
      const int c0 = cb + h8;
      *(v8h*)&Aq[ii][c0] = cvt8(&Qb[(size_t)c0 * HWDIM + pA0 + ii], HWDIM);
    }
  }
  __syncthreads();

  float rmax[8], msum[8];
  #pragma unroll
  for (int v = 0; v < 8; ++v) { rmax[v] = -3.0e38f; msum[v] = 0.0f; }

  for (int j0 = 0; j0 < ROWS; j0 += 64) {            // 45 col supertiles
    const int shotB = j0 / HWDIM;
    const int pB0   = j0 % HWDIM;
    const float* Kb = XK + (size_t)(n * NSHOT + shotB) * PERIMG;
    v8f acc0 = {}, acc1 = {}, acc2 = {}, acc3 = {};
    for (int k0 = 0; k0 < CDIM; k0 += 32) {
      __syncthreads();
      { // stage B: each thread owns column jj=tid, 4 packed b128 stores
        const float* kb0 = &Kb[(size_t)k0 * HWDIM + pB0 + tid];
        #pragma unroll
        for (int kh = 0; kh < 32; kh += 8)
          *(v8h*)&Bs[tid][kh] = cvt8(kb0 + (size_t)kh * HWDIM, HWDIM);
      }
      if (k0 + 32 < CDIM)
        __builtin_prefetch(&Kb[(size_t)(k0 + 32) * HWDIM + pB0 + tid], 0, 1);
      __syncthreads();
      union { v16h v; v8h h[2]; } af;
      const _Float16* ap = &Aq[wv * 16 + r][k0];
      af.h[0] = *(const v8h*)(ap + g * 8);
      af.h[1] = *(const v8h*)(ap + 16 + g * 8);
      #pragma unroll
      for (int t = 0; t < 4; ++t) {
        union { v16h v; v8h h[2]; } bf;
        const _Float16* bp = &Bs[t * 16 + r][0];
        bf.h[0] = *(const v8h*)(bp + g * 8);
        bf.h[1] = *(const v8h*)(bp + 16 + g * 8);
        v8f& acc = (t==0)?acc0:((t==1)?acc1:((t==2)?acc2:acc3));
        acc = __builtin_amdgcn_wmma_f32_16x16x32_f16(
            false, af.v, false, bf.v, (short)0, acc, false, false);
      }
    }
    #pragma unroll
    for (int v = 0; v < 8; ++v)
      rmax[v] = fmaxf(rmax[v],
                fmaxf(fmaxf(acc0[v], acc1[v]), fmaxf(acc2[v], acc3[v])));
    if (((j0 / 64) % 9) == 8) {   // chunk (576 cols) done: cross-lane max
      #pragma unroll
      for (int v = 0; v < 8; ++v) {
        float m = rmax[v];
        m = fmaxf(m, __shfl_xor(m, 1, 32));
        m = fmaxf(m, __shfl_xor(m, 2, 32));
        m = fmaxf(m, __shfl_xor(m, 4, 32));
        m = fmaxf(m, __shfl_xor(m, 8, 32));
        msum[v] += m;
        rmax[v] = -3.0e38f;
      }
    }
  }
  if (r == 0) {
    #pragma unroll
    for (int v = 0; v < 8; ++v)
      xw[n * ROWS + i0 + wv * 16 + g * 8 + v] = msum[v] * 0.2f;
  }
}

// ---------- per-(image,pixel) inverse channel L2 norm ----------
__global__ void k_rownorm_inv(const float* __restrict__ X, float* __restrict__ invn)
{
  int b = blockIdx.y;
  int p = blockIdx.x * blockDim.x + threadIdx.x;
  const float* Xb = X + (size_t)b * PERIMG;
  float s = 0.f;
  for (int c = 0; c < CDIM; ++c) { float v = Xb[(size_t)c * HWDIM + p]; s += v * v; }
  invn[b * HWDIM + p] = 1.0f / fmaxf(sqrtf(s), 1e-12f);
}

// ---------- per-row (25 x 576) max -> argmax mask (softmax monotone) ----------
__global__ void k_rowmax_mask(const float* __restrict__ xw, float* __restrict__ mask)
{
  int row = blockIdx.x, t = threadIdx.x;
  __shared__ float red[192];
  const float* x = xw + row * HWDIM;
  float m = -3.0e38f;
  for (int j = t; j < HWDIM; j += 192) m = fmaxf(m, x[j]);
  red[t] = m; __syncthreads();
  for (int s = 96; s > 0; s >>= 1) { if (t < s) red[t] = fmaxf(red[t], red[t + s]); __syncthreads(); }
  float rm = red[0];
  for (int j = t; j < HWDIM; j += 192) mask[row * HWDIM + j] = (x[j] == rm) ? 1.f : 0.f;
}

// ---------- out[b,c] = scale * sum_p x[b,c,p]*invn[b,p]*(mask?:1) ----------
__global__ __launch_bounds__(640) void k_chansum(
    const float* __restrict__ X, const float* __restrict__ invn,
    const float* __restrict__ msk, float* __restrict__ out, float scale)
{
  int b = blockIdx.x, c = threadIdx.x;
  __shared__ float w[HWDIM];
  for (int p = threadIdx.x; p < HWDIM; p += 640)
    w[p] = invn[b * HWDIM + p] * (msk ? msk[b * HWDIM + p] : 1.0f);
  __syncthreads();
  const float* xb = X + (size_t)b * PERIMG + (size_t)c * HWDIM;
  float acc = 0.f;
  for (int p = 0; p < HWDIM; ++p) acc += xb[p] * w[p];
  out[b * CDIM + c] = acc * scale;
}

// ---------- cor[n,o,p] = sum_k invn[nk,p] * sum_c x5[nk,c,p]*seeds[o,k,c] ----------
__global__ __launch_bounds__(64) void k_cor(
    const float* __restrict__ x5, const float* __restrict__ invn,
    const float* __restrict__ seeds, float* __restrict__ cor)
{
  int n = blockIdx.y;
  int p = blockIdx.x * 64 + threadIdx.x;
  __shared__ float sd[NIMG * CDIM]; // 64000B
  for (int e = threadIdx.x; e < NIMG * CDIM; e += 64) sd[e] = seeds[e];
  __syncthreads();
  float acc[NWAY] = {};
  for (int k = 0; k < NSHOT; ++k) {
    int bi = n * NSHOT + k;
    float iv = invn[bi * HWDIM + p];
    float tk[NWAY] = {};
    const float* xb = x5 + (size_t)bi * PERIMG + p;
    for (int c = 0; c < CDIM; ++c) {
      float xv = xb[(size_t)c * HWDIM];
      #pragma unroll
      for (int o = 0; o < NWAY; ++o) tk[o] += xv * sd[(o * NSHOT + k) * CDIM + c];
    }
    #pragma unroll
    for (int o = 0; o < NWAY; ++o) acc[o] += iv * tk[o];
  }
  for (int o = 0; o < NWAY; ++o) cor[(n * NWAY + o) * HWDIM + p] = acc[o];
}

// ---------- per-row min/max of cor (25 rows) ----------
__global__ void k_minmax(const float* __restrict__ x, float* __restrict__ mm)
{
  int row = blockIdx.x, t = threadIdx.x;
  __shared__ float rmn[192], rmx[192];
  float mn = 3.0e38f, mx = -3.0e38f;
  for (int j = t; j < HWDIM; j += 192) {
    float v = x[row * HWDIM + j]; mn = fminf(mn, v); mx = fmaxf(mx, v);
  }
  rmn[t] = mn; rmx[t] = mx; __syncthreads();
  for (int s = 96; s > 0; s >>= 1) {
    if (t < s) { rmn[t] = fminf(rmn[t], rmn[t+s]); rmx[t] = fmaxf(rmx[t], rmx[t+s]); }
    __syncthreads();
  }
  if (t == 0) { mm[row * 2] = rmn[0]; mm[row * 2 + 1] = rmx[0]; }
}

// ---------- proto[n,c] = mean_{o,p} x5[no,c,p]*cormap[no,p] ----------
__global__ __launch_bounds__(640) void k_proto(
    const float* __restrict__ x5, const float* __restrict__ cor,
    const float* __restrict__ cmm, float* __restrict__ proto)
{
  int n = blockIdx.x, c = threadIdx.x;
  __shared__ float cmap[NSHOT * HWDIM];
  for (int e = threadIdx.x; e < NSHOT * HWDIM; e += 640) {
    int o = e / HWDIM, p = e % HWDIM;
    int row = n * NSHOT + o;
    float mn = cmm[row * 2], mx = cmm[row * 2 + 1];
    cmap[e] = (cor[row * HWDIM + p] - mn) / (mx - mn + 1e-12f);
  }
  __syncthreads();
  float acc = 0.f;
  for (int o = 0; o < NSHOT; ++o) {
    const float* xb = x5 + (size_t)(n * NSHOT + o) * PERIMG + (size_t)c * HWDIM;
    const float* cm = &cmap[o * HWDIM];
    for (int p = 0; p < HWDIM; ++p) acc += xb[p] * cm[p];
  }
  proto[n * CDIM + c] = acc / (float)ROWS;
}

// ---------- cds = sigmoid(d_intra/d_inter) ----------
__global__ void k_cds(const float* __restrict__ lf, const float* __restrict__ invn,
                      const float* __restrict__ Simg, float* __restrict__ cds)
{
  int b = blockIdx.y;
  int p = blockIdx.x * 192 + threadIdx.x;
  int n = b / NSHOT;
  __shared__ float sway[NWAY * CDIM], stot[CDIM], sb[CDIM];
  for (int c = threadIdx.x; c < CDIM; c += 192) {
    float tot = 0.f;
    for (int nn = 0; nn < NWAY; ++nn) {
      float s = 0.f;
      for (int k = 0; k < NSHOT; ++k) s += Simg[(nn * NSHOT + k) * CDIM + c];
      sway[nn * CDIM + c] = s; tot += s;
    }
    stot[c] = tot;
    sb[c]   = Simg[b * CDIM + c];
  }
  __syncthreads();
  float iv = invn[b * HWDIM + p];
  const float* xb = lf + (size_t)b * PERIMG + p;
  float a1 = 0, a2 = 0, a3 = 0, a4 = 0;
  for (int c = 0; c < CDIM; ++c) {
    float x = xb[(size_t)c * HWDIM] * iv;
    a1 += x * sway[n * CDIM + c];
    a2 += x * x;
    a3 += x * stot[c];
    a4 += x * sb[c];
  }
  float d_intra = (a1 - a2) / (float)ROWS;
  float d_inter = (a3 - a4) / (float)(NIMG * HWDIM);
  float z = d_intra / d_inter;
  cds[b * HWDIM + p] = 1.0f / (1.0f + expf(-z));
}

// ---------- stable top-k via O(n^2) rank (k=2304 of 2880 per way) ----------
__global__ __launch_bounds__(512) void k_topk(const float* __restrict__ cds,
                                              float* __restrict__ sel)
{
  int n = blockIdx.x;
  __shared__ float v[ROWS];
  for (int e = threadIdx.x; e < ROWS; e += 512) v[e] = cds[n * ROWS + e];
  __syncthreads();
  for (int i = threadIdx.x; i < ROWS; i += 512) {
    float vi = v[i];
    int cnt = 0;
    for (int j = 0; j < ROWS; ++j) {
      float vj = v[j];
      cnt += (vj > vi) || ((vj == vi) && (j < i));
    }
    sel[n * ROWS + i] = (cnt < TOPK) ? 1.0f : 0.0f;
  }
}

// ---------- pos_index broadcast (the 36.8MB output write) ----------
__global__ void k_broadcast(const float* __restrict__ sel, float* __restrict__ out)
{
  unsigned idx = blockIdx.x * 256u + threadIdx.x;
  unsigned b = idx / PERIMG;
  unsigned p = idx % HWDIM;   // valid: PERIMG % 576 == 0
  out[idx] = sel[b * HWDIM + p];
}

// ---------- contrastive loss from m2 (25x640) ----------
__global__ __launch_bounds__(256) void k_losses(const float* __restrict__ m2,
                                                float* __restrict__ out0)
{
  __shared__ float m1[NWAY * CDIM];
  __shared__ float red[256];
  int t = threadIdx.x;
  for (int c = t; c < CDIM; c += 256)
    for (int n = 0; n < NWAY; ++n) {
      float s = 0.f;
      for (int k = 0; k < NSHOT; ++k) s += m2[(n * NSHOT + k) * CDIM + c];
      m1[n * CDIM + c] = s * 0.2f;
    }
  __syncthreads();
  float pin = 0.f, pout = 0.f;
  for (int c = t; c < CDIM; c += 256) {
    for (int a = 0; a < NWAY; ++a)
      for (int b2 = 0; b2 < NWAY; ++b2)
        if (a != b2) pin += m1[a * CDIM + c] * m1[b2 * CDIM + c];
    for (int n = 0; n < NWAY; ++n)
      for (int a = 0; a < NSHOT; ++a)
        for (int b2 = 0; b2 < NSHOT; ++b2)
          if (a != b2)
            pout += m2[(n * NSHOT + a) * CDIM + c] * m2[(n * NSHOT + b2) * CDIM + c];
  }
  red[t] = pin; __syncthreads();
  for (int s = 128; s > 0; s >>= 1) { if (t < s) red[t] += red[t + s]; __syncthreads(); }
  float intra_sum = red[0]; __syncthreads();
  red[t] = pout; __syncthreads();
  for (int s = 128; s > 0; s >>= 1) { if (t < s) red[t] += red[t + s]; __syncthreads(); }
  if (t == 0) {
    float me = red[0] / 125.0f;      // l_inter.mean()
    float mi = intra_sum / 25.0f;    // l_intra.mean()
    out0[0] = expf(me / mi);
  }
}

__global__ void k_init(float* a) { if (threadIdx.x < 2) a[threadIdx.x] = 0.f; }

// ---------- CTC exp-sum accumulation ----------
__global__ void k_ctc(const float* __restrict__ lf, const float* __restrict__ invn,
                      const float* __restrict__ sel, const float* __restrict__ proto,
                      float* __restrict__ acc)
{
  int b = blockIdx.y;
  int p = blockIdx.x * 192 + threadIdx.x;
  int n = b / NSHOT;
  float iv = invn[b * HWDIM + p];
  float sv = sel[b * HWDIM + p];
  const float* xb = lf + (size_t)b * PERIMG + p;
  const float* g = proto + n * CDIM;
  float s1 = 0.f, s2 = 0.f;
  for (int c = 0; c < CDIM; ++c) {
    float lfn = xb[(size_t)c * HWDIM] * iv;
    float gv = g[c];
    s1 += expf(sv * lfn * gv);
    s2 += expf((1.0f - sv) * lfn * gv);
  }
  __shared__ float r1[192], r2[192];
  int t = threadIdx.x;
  r1[t] = s1; r2[t] = s2; __syncthreads();
  for (int s = 96; s > 0; s >>= 1) {
    if (t < s) { r1[t] += r1[t + s]; r2[t] += r2[t + s]; }
    __syncthreads();
  }
  if (t == 0) { atomicAdd(&acc[0], r1[0]); atomicAdd(&acc[1], r2[0]); }
}

__global__ void k_final(const float* __restrict__ acc, float* __restrict__ out1)
{
  out1[0] = -logf(acc[0] / acc[1]);
}

// =====================================================================
extern "C" void kernel_launch(void* const* d_in, const int* in_sizes, int n_in,
                              void* d_out, int out_size, void* d_ws, size_t ws_size,
                              hipStream_t stream) {
  const float* lf = (const float*)d_in[0];
  const float* Wc = (const float*)d_in[1];
  const float* bc = (const float*)d_in[2];
  const float* Wq = (const float*)d_in[3];
  const float* bq = (const float*)d_in[4];
  const float* Wk = (const float*)d_in[5];
  const float* bk = (const float*)d_in[6];
  float* out = (float*)d_out;
  float* ws  = (float*)d_ws;

  const size_t TOT = (size_t)NIMG * PERIMG; // 9,216,000 floats
  float* B1 = ws;               // x5a, later reused for xk
  float* B2 = ws + TOT;         // x5
  float* B3 = ws + 2 * TOT;     // xq
  size_t S = 3 * TOT;
  float* xw      = ws + S;            S += NWAY * ROWS;     // 14400
  float* invn_x5 = ws + S;            S += NIMG * HWDIM;    // 14400
  float* attmask = ws + S;            S += NIMG * HWDIM;
  float* seeds   = ws + S;            S += NIMG * CDIM;     // 16000
  float* cor     = ws + S;            S += NIMG * HWDIM;
  float* cmm     = ws + S;            S += 64;
  float* proto   = ws + S;            S += NWAY * CDIM;
  float* invn_lf = ws + S;            S += NIMG * HWDIM;
  float* Simg    = ws + S;            S += NIMG * CDIM;
  float* cds     = ws + S;            S += NIMG * HWDIM;
  float* sel     = ws + S;            S += NIMG * HWDIM;
  float* m2      = ws + S;            S += NIMG * CDIM;
  float* accp    = ws + S;            S += 2;

  dim3 gG(5, 9, NIMG);
  // conv chain (WMMA f16 GEMMs)
  k_conv1x1<true ><<<gG, 256, 0, stream>>>(Wc, bc, lf, lf,      B1); // x5a = conv(x,Wc)+x
  k_conv1x1<false><<<gG, 256, 0, stream>>>(Wq, bq, B1, nullptr, B2); // x5
  k_conv1x1<false><<<gG, 256, 0, stream>>>(Wq, bq, B2, nullptr, B3); // xq
  k_conv1x1<false><<<gG, 256, 0, stream>>>(Wk, bk, B2, nullptr, B1); // xk (reuse B1)

  // fused attention row statistic (WMMA)
  k_attn_rowstat<<<dim3(ROWS / 32, NWAY), 64, 0, stream>>>(B3, B1, xw);

  // attlayer tail
  k_rownorm_inv<<<dim3(3, NIMG), 192, 0, stream>>>(B2, invn_x5);
  k_rowmax_mask<<<NIMG, 192, 0, stream>>>(xw, attmask);
  k_chansum<<<NIMG, 640, 0, stream>>>(B2, invn_x5, attmask, seeds, 1.0f);
  k_cor<<<dim3(9, NWAY), 64, 0, stream>>>(B2, invn_x5, seeds, cor);
  k_minmax<<<NIMG, 192, 0, stream>>>(cor, cmm);
  k_proto<<<NWAY, 640, 0, stream>>>(B2, cor, cmm, proto);

  // cdsnet
  k_rownorm_inv<<<dim3(3, NIMG), 192, 0, stream>>>(lf, invn_lf);
  k_chansum<<<NIMG, 640, 0, stream>>>(lf, invn_lf, nullptr, Simg, 1.0f);
  k_cds<<<dim3(3, NIMG), 192, 0, stream>>>(lf, invn_lf, Simg, cds);
  k_topk<<<NWAY, 512, 0, stream>>>(cds, sel);
  k_broadcast<<<(unsigned)(TOT / 256), 256, 0, stream>>>(sel, out + 2);
  k_chansum<<<NIMG, 640, 0, stream>>>(lf, invn_lf, sel, m2, 1.0f / (float)HWDIM);
  k_losses<<<1, 256, 0, stream>>>(m2, out + 0);

  // ctc loss
  k_init<<<1, 32, 0, stream>>>(accp);
  k_ctc<<<dim3(3, NIMG), 192, 0, stream>>>(lf, invn_lf, sel, proto, accp);
  k_final<<<1, 1, 0, stream>>>(accp, out + 1);
}
